// DFDB_17136919511807
// MI455X (gfx1250) — matrix operand order repaired and checked by reading, not compiled, not run on hardware
//
#include <hip/hip_runtime.h>
#include <hip/hip_bf16.h>
#include <math.h>

// ---------------- problem constants ----------------
#define BB   8
#define TT   288
#define NN   2000
#define FL   145
#define FLP  160      // FL padded to multiple of 32
#define EE   128
#define IDD  16
#define DD   144      // E + ID
#define DDP  160      // padded to multiple of 32
#define HH   128
#define KK   20
#define MROWS (BB*NN) // 16000
#define TWC  320      // twiddle cols: [0,145)=cos, [160,305)=sin, rest 0

typedef __attribute__((ext_vector_type(16))) _Float16 v16h;
typedef __attribute__((ext_vector_type(8)))  float    v8f;
typedef int vi4 __attribute__((vector_size(16)));   // matches async builtin param type

union AF { v16h v; _Float16 h[16]; unsigned u[8]; };
union CF { v8f  v; float    f[8];  };

#define AS1 __attribute__((address_space(1)))
#define AS3 __attribute__((address_space(3)))

// gfx1250 async global->LDS copy, guarded so compilation can never break.
#if defined(__HIP_DEVICE_COMPILE__)
#  if __has_builtin(__builtin_amdgcn_global_load_async_to_lds_b128)
#    define USE_ASYNC_LDS 1
#  endif
#endif

// A-fragment fill from two contiguous 16B chunks
__device__ __forceinline__ void fill_a(AF& a, uint4 c1, uint4 c2) {
  a.u[0] = c1.x; a.u[1] = c1.y; a.u[2] = c1.z; a.u[3] = c1.w;
  a.u[4] = c2.x; a.u[5] = c2.y; a.u[6] = c2.z; a.u[7] = c2.w;
}

// ---------------- 0a) LDS-tiled transpose: x[B,T,N] f32 -> xth[B*N, T] f16 ----------------
__global__ __launch_bounds__(256) void k_transpose_x(const float* __restrict__ x,
                                                     _Float16* __restrict__ xth) {
  const int b  = blockIdx.x;
  const int t0 = blockIdx.y * 32;
  const int n0 = blockIdx.z * 32;
  __shared__ float tile[32][33];
#pragma unroll
  for (int r = 0; r < 4; ++r) {
    const int idx = r * 256 + threadIdx.x;
    const int tt = idx >> 5, nn = idx & 31;
    const int n = n0 + nn;
    tile[tt][nn] = (n < NN) ? x[((size_t)b * TT + t0 + tt) * NN + n] : 0.f;
  }
  __syncthreads();
#pragma unroll
  for (int r = 0; r < 4; ++r) {
    const int idx = r * 256 + threadIdx.x;
    const int nn = idx >> 5, tt = idx & 31;
    const int n = n0 + nn;
    if (n < NN)
      xth[((size_t)b * NN + n) * TT + t0 + tt] = (_Float16)tile[tt][nn];
  }
}

// ---------------- 0b) twiddle pack: WC[j][t] col-major f16 ----------------
__global__ __launch_bounds__(256) void k_pack_tw(_Float16* __restrict__ WC) {
  const int idx = blockIdx.x * 256 + threadIdx.x;   // TWC*TT total
  if (idx >= TWC * TT) return;
  const int j = idx / TT, t = idx % TT;
  const float c = -6.28318530717958647692f / (float)TT;
  float v = 0.f;
  if (j < FL) {                                     // cos plane
    float sv, cv; __sincosf(c * (float)((j * t) % TT), &sv, &cv);
    v = cv;
  } else if (j >= FLP && j < FLP + FL) {            // sin plane
    const int k = j - FLP;
    float sv, cv; __sincosf(c * (float)((k * t) % TT), &sv, &cv);
    v = sv;
  }
  WC[idx] = (_Float16)v;
}

// ---------------- 0c) WMMA GEMM: dftr = xth[MROWS,288] @ WC^T -> [MROWS,TWC] f32 ----------------
__global__ __launch_bounds__(256) void k_gemm_dft(const _Float16* __restrict__ xth,
                                                  const _Float16* __restrict__ WC,
                                                  float* __restrict__ dftr) {
  const int mt = blockIdx.x;                 // 0..999
  const int coltile = blockIdx.y * 8 + (threadIdx.x >> 5);  // 0..23, valid < 20
  if (coltile >= TWC / 16) return;
  const int lane = threadIdx.x & 31;
  const int g = lane >> 4, lr = lane & 15;
  const int m0 = mt * 16;
  const int row = m0 + lr;
  const int col = coltile * 16 + lr;
  CF c; c.v = (v8f){};
  const uint4*    sa = (const uint4*)(xth + (size_t)row * TT);
  const _Float16* bp = WC + (size_t)col * TT;
  for (int k0 = 0; k0 < TT; k0 += 32) {      // 9 WMMAs
    AF a, bf;
    fill_a(a, sa[(k0 + 16 * g) >> 3], sa[((k0 + 16 * g) >> 3) + 1]);
    bf.v = *(const v16h*)(bp + k0 + 16 * g);
    c.v = __builtin_amdgcn_wmma_f32_16x16x32_f16(false, a.v, false, bf.v,
                                                 (short)0, c.v, false, false);
  }
#pragma unroll
  for (int r = 0; r < 8; ++r)
    dftr[(size_t)(m0 + r + 8 * g) * TWC + col] = c.f[r];
}

// ---------------- 0d) magnitude: xf[row,k] = sqrt(re^2 + im^2) ----------------
__global__ __launch_bounds__(160) void k_mag(const float* __restrict__ dftr,
                                             float* __restrict__ xf) {
  const int rid = blockIdx.x;
  const int k = threadIdx.x;
  if (k < FL) {
    const float re = dftr[(size_t)rid * TWC + k];
    const float im = dftr[(size_t)rid * TWC + FLP + k];
    xf[(size_t)rid * FL + k] = sqrtf(re * re + im * im);
  }
}

// ---------------- 2a) node-norm partial sums (coalesced), atomic accumulate ----------------
__global__ __launch_bounds__(160) void k_colnorm_part(const float* __restrict__ xf,
                                                      float* __restrict__ s1acc) {
  const int b = blockIdx.x;                  // 0..7
  const int chunk = blockIdx.y;              // 0..15, 125 n's each
  const int f = threadIdx.x;                 // 0..159
  float acc = 0.f;
  const int nbeg = chunk * 125, nend = nbeg + 125;
  if (f < FL) {
    for (int n = nbeg; n < nend; ++n) {
      const float v = xf[((size_t)b * NN + n) * FL + f];   // lanes: consecutive f
      acc += v * v;
    }
    atomicAdd(&s1acc[b * FL + f], acc);
  }
}

// ---------------- 2b) finalize s1 ----------------
__global__ __launch_bounds__(256) void k_colfin(const float* __restrict__ s1acc,
                                                float* __restrict__ s1) {
  const int idx = blockIdx.x * 256 + threadIdx.x;
  if (idx < BB * FL) s1[idx] = fmaxf(sqrtf(s1acc[idx]), 1e-12f);
}

// ---------------- 3) row-normalize + pack f16 (padded to FLP) ----------------
__global__ __launch_bounds__(160) void k_rownorm(const float* __restrict__ xf,
                                                 const float* __restrict__ s1,
                                                 _Float16* __restrict__ xfn) {
  const int rid = blockIdx.x;            // b*NN + n
  const int b = rid / NN;
  const int f = threadIdx.x;             // 0..159
  __shared__ float sd[160];
  float v = 0.f;
  if (f < FL) v = xf[(size_t)rid * FL + f] / s1[b * FL + f];
  sd[f] = v * v;
  __syncthreads();
  for (int sft = 80; sft >= 5; sft >>= 1) {
    if (f < sft) sd[f] += sd[f + sft];
    __syncthreads();
  }
  if (f == 0) sd[0] = fmaxf(sqrtf(sd[0] + sd[1] + sd[2] + sd[3] + sd[4]), 1e-12f);
  __syncthreads();
  const float rn = sd[0];
  xfn[(size_t)rid * FLP + f] = (f < FL) ? (_Float16)(v / rn) : (_Float16)0.f;
}

// ---------------- 4a) pack Ex -> f16 COLUMN-MAJOR [EE][FLP] ----------------
__global__ __launch_bounds__(256) void k_pack_ex(const float* __restrict__ Ex,
                                                 _Float16* __restrict__ ExC) {
  const int idx = blockIdx.x * 256 + threadIdx.x;   // 0 .. EE*FLP-1
  if (idx >= EE * FLP) return;
  const int j = idx / FLP, k = idx % FLP;
  ExC[idx] = (k < FL) ? (_Float16)Ex[k * EE + j] : (_Float16)0.f;
}

// ---------------- 4b) pack Wxabs -> f16 COLUMN-MAJOR [HH][HH] ----------------
__global__ __launch_bounds__(256) void k_pack_wx(const float* __restrict__ Wx,
                                                 _Float16* __restrict__ WxC) {
  const int idx = blockIdx.x * 256 + threadIdx.x;   // 0 .. HH*HH-1
  if (idx >= HH * HH) return;
  const int j = idx / HH, k = idx % HH;
  WxC[idx] = (_Float16)Wx[k * HH + j];
}

// ---------------- 4c) pack node ids into xk tail (cols 128..159) ----------------
__global__ __launch_bounds__(256) void k_pack_nodes(const float* __restrict__ nodes,
                                                    _Float16* __restrict__ xk) {
  const int idx = blockIdx.x * 256 + threadIdx.x;   // MROWS*32 total
  if (idx >= MROWS * 32) return;
  const int row = idx / 32, j = idx % 32;
  const int n = row % NN;
  xk[(size_t)row * DDP + EE + j] = (j < IDD) ? (_Float16)nodes[n * IDD + j]
                                             : (_Float16)0.f;
}

// ---------------- 5) WMMA GEMM: xe = xfn[MROWS,FLP] @ Ex -> xk[:,0:128] ----------------
__global__ __launch_bounds__(256) void k_gemm_embed(const _Float16* __restrict__ xfn,
                                                    const _Float16* __restrict__ ExC,
                                                    _Float16* __restrict__ xk) {
  const int mt = blockIdx.x;                 // 0..999
  const int wave = threadIdx.x >> 5;         // 8 col-tiles
  const int lane = threadIdx.x & 31;
  const int g = lane >> 4, lr = lane & 15;
  const int m0 = mt * 16;
  const int row = m0 + lr;
  const int col = wave * 16 + lr;
  CF c; c.v = (v8f){};
  const uint4*    sa = (const uint4*)(xfn + (size_t)row * FLP);
  const _Float16* bp = ExC + (size_t)col * FLP;
  for (int k0 = 0; k0 < FLP; k0 += 32) {
    AF a, bf;
    fill_a(a, sa[(k0 + 16 * g) >> 3], sa[((k0 + 16 * g) >> 3) + 1]);
    bf.v = *(const v16h*)(bp + k0 + 16 * g);
    c.v = __builtin_amdgcn_wmma_f32_16x16x32_f16(false, a.v, false, bf.v,
                                                 (short)0, c.v, false, false);
  }
#pragma unroll
  for (int r = 0; r < 8; ++r) {
    const int mrow = m0 + r + 8 * g;
    xk[(size_t)mrow * DDP + col] = (_Float16)c.f[r];
  }
}

// ---------------- 6) per-node WMMA GEMM: x1 = relu(xk[:,n,:] @ Wd[n]) ----------------
__global__ __launch_bounds__(256) void k_gemm_node(const _Float16* __restrict__ xk,
                                                   const float* __restrict__ Wd,
                                                   float* __restrict__ x1,
                                                   _Float16* __restrict__ x1h) {
  const int n = blockIdx.x;                  // node 0..1999
  __shared__ _Float16 shWd[HH * DDP];        // [col][k], 40 KB
  const float* wsrc = Wd + (size_t)n * DD * HH;
  for (int i = threadIdx.x; i < DD * HH; i += 256) {
    const int k = i / HH, col = i % HH;
    shWd[col * DDP + k] = (_Float16)wsrc[i];
  }
  for (int i = threadIdx.x; i < HH * (DDP - DD); i += 256) {
    const int col = i / (DDP - DD), k = DD + i % (DDP - DD);
    shWd[col * DDP + k] = (_Float16)0.f;
  }
  __syncthreads();

  const int wave = threadIdx.x >> 5;
  const int lane = threadIdx.x & 31;
  const int g = lane >> 4, lr = lane & 15;
  const int col = wave * 16 + lr;
  CF c; c.v = (v8f){};
  const uint4* sa = (const uint4*)(xk + ((size_t)lr * NN + n) * DDP); // valid if lr<BB
  for (int k0 = 0; k0 < DDP; k0 += 32) {
    AF a, bf;
    if (lr < BB) {
      fill_a(a, sa[(k0 + 16 * g) >> 3], sa[((k0 + 16 * g) >> 3) + 1]);
    } else {
#pragma unroll
      for (int p = 0; p < 8; ++p) a.u[p] = 0u;
    }
    bf.v = *(const v16h*)(shWd + (size_t)col * DDP + k0 + 16 * g);
    c.v = __builtin_amdgcn_wmma_f32_16x16x32_f16(false, a.v, false, bf.v,
                                                 (short)0, c.v, false, false);
  }
#pragma unroll
  for (int r = 0; r < 8; ++r) {
    const int m = r + 8 * g;                 // batch index
    if (m < BB) {
      const float v = fmaxf(c.f[r], 0.f);
      const size_t idx = ((size_t)m * NN + n) * HH + col;
      x1[idx]  = v;
      x1h[idx] = (_Float16)v;
    }
  }
}

// ---------------- 7) LayerNorm stats over (N,H) per batch ----------------
__global__ __launch_bounds__(256) void k_lnstats(const float* __restrict__ x1,
                                                 float* __restrict__ stats) {
  const int b = blockIdx.x;
  const float* p = x1 + (size_t)b * NN * HH;
  __shared__ float sA[256], sB[256];
  float s = 0.f, ss = 0.f;
  for (int i = threadIdx.x; i < NN * HH; i += 256) {
    const float v = p[i];
    s += v; ss += v * v;
  }
  sA[threadIdx.x] = s; sB[threadIdx.x] = ss;
  __syncthreads();
  for (int sft = 128; sft > 0; sft >>= 1) {
    if (threadIdx.x < sft) {
      sA[threadIdx.x] += sA[threadIdx.x + sft];
      sB[threadIdx.x] += sB[threadIdx.x + sft];
    }
    __syncthreads();
  }
  if (threadIdx.x == 0) {
    const float inv = 1.f / (float)(NN * HH);
    const float mu = sA[0] * inv;
    const float var = sB[0] * inv - mu * mu;
    stats[2 * b]     = mu;
    stats[2 * b + 1] = rsqrtf(var + 1e-8f);
  }
}

// ---------------- 8) WMMA GEMM: adp = LN(x1) @ Wxabs (f16 out) ----------------
__global__ __launch_bounds__(256) void k_gemm_adp(const float* __restrict__ x1,
                                                  const _Float16* __restrict__ WxC,
                                                  const float* __restrict__ stats,
                                                  _Float16* __restrict__ adph) {
  const int mt = blockIdx.x;                 // 0..999 (NN%16==0 so no batch straddle)
  const int wave = threadIdx.x >> 5;
  const int lane = threadIdx.x & 31;
  const int g = lane >> 4, lr = lane & 15;
  const int m0 = mt * 16;
  const int row = m0 + lr;
  const int b = row / NN;
  const float mu = stats[2 * b], rs = stats[2 * b + 1];
  const int col = wave * 16 + lr;
  CF c; c.v = (v8f){};
  const float4*   fa = (const float4*)(x1 + (size_t)row * HH);
  const _Float16* bp = WxC + (size_t)col * HH;
  for (int k0 = 0; k0 < HH; k0 += 32) {
    AF a, bf;
    const float4 q0 = fa[((k0 + 16 * g) >> 2)];
    const float4 q1 = fa[((k0 + 16 * g) >> 2) + 1];
    const float4 q2 = fa[((k0 + 16 * g) >> 2) + 2];
    const float4 q3 = fa[((k0 + 16 * g) >> 2) + 3];
    a.h[0]  = (_Float16)((q0.x - mu) * rs); a.h[1]  = (_Float16)((q0.y - mu) * rs);
    a.h[2]  = (_Float16)((q0.z - mu) * rs); a.h[3]  = (_Float16)((q0.w - mu) * rs);
    a.h[4]  = (_Float16)((q1.x - mu) * rs); a.h[5]  = (_Float16)((q1.y - mu) * rs);
    a.h[6]  = (_Float16)((q1.z - mu) * rs); a.h[7]  = (_Float16)((q1.w - mu) * rs);
    a.h[8]  = (_Float16)((q2.x - mu) * rs); a.h[9]  = (_Float16)((q2.y - mu) * rs);
    a.h[10] = (_Float16)((q2.z - mu) * rs); a.h[11] = (_Float16)((q2.w - mu) * rs);
    a.h[12] = (_Float16)((q3.x - mu) * rs); a.h[13] = (_Float16)((q3.y - mu) * rs);
    a.h[14] = (_Float16)((q3.z - mu) * rs); a.h[15] = (_Float16)((q3.w - mu) * rs);
    bf.v = *(const v16h*)(bp + k0 + 16 * g);
    c.v = __builtin_amdgcn_wmma_f32_16x16x32_f16(false, a.v, false, bf.v,
                                                 (short)0, c.v, false, false);
  }
#pragma unroll
  for (int r = 0; r < 8; ++r)
    adph[(size_t)(m0 + r + 8 * g) * HH + col] = (_Float16)c.f[r];
}

// ---------------- 9) WMMA GEMM: adj[b,n,m] = relu(sum_h adp[b,n,h]*x1[b,m,h]) ----------------
__global__ __launch_bounds__(256) void k_gemm_adj(const _Float16* __restrict__ adph,
                                                  const _Float16* __restrict__ x1h,
                                                  float* __restrict__ out) {
  const int nt = blockIdx.x;                 // 0..15  (col tile of 128)
  const int mt = blockIdx.y;                 // 0..124 (row tile of 16)
  const int b  = blockIdx.z;                 // 0..7
  const int m0 = mt * 16;
  __shared__ _Float16 shA[16 * HH];          // 4 KB

  const int wave = threadIdx.x >> 5;
  const int lane = threadIdx.x & 31;
  const int g = lane >> 4, lr = lane & 15;
  const int col = nt * 128 + wave * 16 + lr;
  const bool colok = (col < NN);
  const _Float16* bp = x1h + ((size_t)b * NN + (colok ? col : 0)) * HH;
  if (colok) __builtin_prefetch(bp, 0, 0);   // global_prefetch, overlaps A staging

  {
    const int arow = threadIdx.x >> 4;       // 0..15
    const int achk = threadIdx.x & 15;       // 16B chunk within row
    const _Float16* gsrc = adph + ((size_t)b * NN + m0 + arow) * HH + achk * 8;
    _Float16* ldst = shA + arow * HH + achk * 8;
#ifdef USE_ASYNC_LDS
    __builtin_amdgcn_global_load_async_to_lds_b128(
        (AS1 vi4*)gsrc, (AS3 vi4*)ldst, 0, 0);
#  if __has_builtin(__builtin_amdgcn_s_wait_asynccnt)
    __builtin_amdgcn_s_wait_asynccnt(0);
#  else
    asm volatile("s_wait_asynccnt 0x0" ::: "memory");
#  endif
#else
    *(uint4*)ldst = *(const uint4*)gsrc;
#endif
  }
  __syncthreads();

  CF c; c.v = (v8f){};
  const uint4* sa = (const uint4*)(shA + (size_t)lr * HH);
  for (int k0 = 0; k0 < HH; k0 += 32) {
    AF a, bf;
    fill_a(a, sa[(k0 + 16 * g) >> 3], sa[((k0 + 16 * g) >> 3) + 1]);
    if (colok) {
      bf.v = *(const v16h*)(bp + k0 + 16 * g);  // 16 contiguous f16 (32B aligned)
    } else {
#pragma unroll
      for (int p = 0; p < 8; ++p) bf.u[p] = 0u;
    }
    c.v = __builtin_amdgcn_wmma_f32_16x16x32_f16(false, a.v, false, bf.v,
                                                 (short)0, c.v, false, false);
  }
  if (colok) {
#pragma unroll
    for (int r = 0; r < 8; ++r) {
      const int mrow = m0 + r + 8 * g;
      out[(size_t)b * NN * NN + (size_t)mrow * NN + col] = fmaxf(c.f[r], 0.f);
    }
  }
}

// ---------------- 10) exact top-K mask + softmax per row, in place ----------------
__global__ __launch_bounds__(256) void k_topk_softmax(float* __restrict__ out) {
  const int rid = blockIdx.x;                // b*NN + n
  const int b = rid / NN, n = rid % NN;
  float* row = out + (size_t)b * NN * NN + (size_t)n * NN;
  __shared__ float work[NN];
  __shared__ float sval[256];
  __shared__ int   sidx[256];
  const int tid = threadIdx.x;
  const float NEG = -3.0e38f;

  for (int m = tid; m < NN; m += 256) work[m] = row[m];
  __syncthreads();

  // K rounds of argmax (tie-break: lowest index, matching lax.top_k)
  for (int it = 0; it < KK; ++it) {
    float bv = NEG; int bi = 0x7fffffff;
    for (int m = tid; m < NN; m += 256) {
      const float v = work[m];
      if (v > bv || (v == bv && m < bi)) { bv = v; bi = m; }
    }
    sval[tid] = bv; sidx[tid] = bi;
    __syncthreads();
    for (int sft = 128; sft > 0; sft >>= 1) {
      if (tid < sft) {
        const float ov = sval[tid + sft]; const int oi = sidx[tid + sft];
        if (ov > sval[tid] || (ov == sval[tid] && oi < sidx[tid])) {
          sval[tid] = ov; sidx[tid] = oi;
        }
      }
      __syncthreads();
    }
    if (tid == 0) work[sidx[0]] = NEG;       // mark selected
    __syncthreads();
  }

  float lmax = 0.f;                          // zeros present in every row
  for (int m = tid; m < NN; m += 256)
    if (work[m] == NEG) lmax = fmaxf(lmax, row[m]);
  sval[tid] = lmax;
  __syncthreads();
  for (int sft = 128; sft > 0; sft >>= 1) {
    if (tid < sft) sval[tid] = fmaxf(sval[tid], sval[tid + sft]);
    __syncthreads();
  }
  const float rmax = sval[0];
  __syncthreads();

  float lsum = 0.f;
  for (int m = tid; m < NN; m += 256) {
    const float z = (work[m] == NEG) ? row[m] : 0.f;
    lsum += __expf(z - rmax);
  }
  sval[tid] = lsum;
  __syncthreads();
  for (int sft = 128; sft > 0; sft >>= 1) {
    if (tid < sft) sval[tid] += sval[tid + sft];
    __syncthreads();
  }
  const float rinv = 1.f / sval[0];
  __syncthreads();

  for (int m = tid; m < NN; m += 256) {
    const float z = (work[m] == NEG) ? row[m] : 0.f;
    row[m] = __expf(z - rmax) * rinv;
  }
}

// ---------------- launch ----------------
extern "C" void kernel_launch(void* const* d_in, const int* in_sizes, int n_in,
                              void* d_out, int out_size, void* d_ws, size_t ws_size,
                              hipStream_t stream) {
  const float* x     = (const float*)d_in[0];  // [B,T,N,1]
  const float* Ex    = (const float*)d_in[1];  // [FL,E]
  const float* nodes = (const float*)d_in[2];  // [N,ID]
  const float* Wd    = (const float*)d_in[3];  // [N,D,H]
  const float* Wxabs = (const float*)d_in[4];  // [H,H]
  float* out = (float*)d_out;                  // [B,N,N]

  // workspace carve-out (~45 MB peak), 256B aligned slices
  char* w = (char*)d_ws;
  auto alloc = [&](size_t bytes) -> char* {
    char* p = w;
    w += (bytes + 255) & ~(size_t)255;
    return p;
  };
  float*    xf    = (float*)   alloc((size_t)MROWS * FL * 4);
  float*    s1    = (float*)   alloc((size_t)BB * FL * 4);
  float*    s1acc = (float*)   alloc((size_t)BB * FL * 4);
  _Float16* xfn   = (_Float16*)alloc((size_t)MROWS * FLP * 2);
  _Float16* ExC   = (_Float16*)alloc((size_t)EE * FLP * 2);
  _Float16* WxC   = (_Float16*)alloc((size_t)HH * HH * 2);
  _Float16* xth   = (_Float16*)alloc((size_t)MROWS * TT * 2);
  _Float16* WC    = (_Float16*)alloc((size_t)TWC * TT * 2);
  // union region: dftr (re/im planes) aliases the xk/x1/x1h/adph group.
  // dftr is dead after k_mag, before any of the aliased buffers are written.
  char* ubase = w;
  _Float16* xk    = (_Float16*)alloc((size_t)MROWS * DDP * 2);
  float*    x1    = (float*)   alloc((size_t)MROWS * HH * 4);
  _Float16* x1h   = (_Float16*)alloc((size_t)MROWS * HH * 2);
  _Float16* adph  = (_Float16*)alloc((size_t)MROWS * HH * 2);
  float*    dftr  = (float*)ubase;             // MROWS*TWC*4 = 20.5 MB (fits in group)
  float*    stats = (float*)   alloc(64);

  // DFT as WMMA GEMM with twiddle matrix
  dim3 gt(BB, TT / 32, (NN + 31) / 32);
  k_transpose_x<<<gt, 256, 0, stream>>>(x, xth);
  k_pack_tw    <<<(TWC * TT + 255) / 256, 256, 0, stream>>>(WC);
  dim3 gd(MROWS / 16, 3);
  k_gemm_dft   <<<gd, 256, 0, stream>>>(xth, WC, dftr);
  k_mag        <<<MROWS, 160, 0, stream>>>(dftr, xf);

  // double normalization
  hipMemsetAsync(s1acc, 0, (size_t)BB * FL * 4, stream);
  dim3 gc(BB, 16);
  k_colnorm_part<<<gc, 160, 0, stream>>>(xf, s1acc);
  k_colfin     <<<(BB * FL + 255) / 256, 256, 0, stream>>>(s1acc, s1);
  k_rownorm    <<<MROWS, 160, 0, stream>>>(xf, s1, xfn);

  // embeds + per-node transform + layernorm + adjacency
  k_pack_ex    <<<(EE * FLP + 255) / 256, 256, 0, stream>>>(Ex, ExC);
  k_pack_wx    <<<(HH * HH + 255) / 256, 256, 0, stream>>>(Wxabs, WxC);
  k_pack_nodes <<<(MROWS * 32 + 255) / 256, 256, 0, stream>>>(nodes, xk);
  k_gemm_embed <<<MROWS / 16, 256, 0, stream>>>(xfn, ExC, xk);
  k_gemm_node  <<<NN, 256, 0, stream>>>(xk, Wd, x1, x1h);
  k_lnstats    <<<BB, 256, 0, stream>>>(x1, stats);
  k_gemm_adp   <<<MROWS / 16, 256, 0, stream>>>(x1, WxC, stats, adph);
  dim3 g4(16, 125, BB);
  k_gemm_adj   <<<g4, 256, 0, stream>>>(adph, x1h, out);
  k_topk_softmax<<<MROWS, 256, 0, stream>>>(out);
}
